// ConvPool_38053410242736
// MI455X (gfx1250) — matrix operand order repaired and checked by reading, not compiled, not run on hardware
//
#include <hip/hip_runtime.h>
#include <hip/hip_bf16.h>

typedef __attribute__((ext_vector_type(16))) __bf16 v16bf;
typedef __attribute__((ext_vector_type(8)))  float  v8f;
typedef __attribute__((ext_vector_type(4)))  float  f32x4;   // native vector for NT stores

constexpr int    Bc   = 64;
constexpr int    Nc   = 1024;
constexpr int    NNc  = Bc * Nc;          // 65536 nodes
constexpr int    Dc   = 128;              // DIN == DOUT
constexpr long long Ec = 1048576;         // 64 * 16384 edges
constexpr int    Kc   = 512;              // kept nodes per graph
constexpr int    BKc  = Bc * Kc;          // 32768 kept rows
constexpr int    WPK_ELEMS = 8 * 4 * 32 * 32;  // fragments(8 nt * 4 kb * 32 lanes) * 32 bf16

// ---------------------------------------------------------------------------
// Generic fills
// ---------------------------------------------------------------------------
__global__ void fill_f32_kernel(float* p, float v, size_t n) {
  size_t i = (size_t)blockIdx.x * blockDim.x + threadIdx.x;
  if (i < n) p[i] = v;
}
__global__ void fill_u32_kernel(unsigned* p, unsigned v, size_t n) {
  size_t i = (size_t)blockIdx.x * blockDim.x + threadIdx.x;
  if (i < n) p[i] = v;
}

// ---------------------------------------------------------------------------
// One-shot weight pre-pack: W[128x128] fp32 -> fragment-ordered bf16 hi/lo.
// Fragment id t = (nt*4 + kbi)*32 + lane holds the 16 B-matrix values this
// lane feeds to v_wmma (K = kbi*32 + (lane>>4)*16 + j, N = nt*16 + (lane&15)),
// stored as [16 hi][16 lo] contiguous (64 B) for two v16bf vector loads.
// ---------------------------------------------------------------------------
__global__ void pack_w_kernel(const float* __restrict__ W, __bf16* __restrict__ Wpk) {
  const int t = blockIdx.x * blockDim.x + threadIdx.x;   // 1024 fragments
  if (t >= 1024) return;
  const int lane = t & 31;
  const int kbi  = (t >> 5) & 3;
  const int nt   = t >> 7;
  const int h = lane >> 4;
  const int n = nt * 16 + (lane & 15);
  const int kb = kbi * 32 + h * 16;
  v16bf hi, lo;
#pragma unroll
  for (int j = 0; j < 16; ++j) {
    float xv = W[(size_t)(kb + j) * Dc + n];
    __bf16 hb = (__bf16)xv;
    hi[j] = hb;
    lo[j] = (__bf16)(xv - (float)hb);
  }
  v16bf* dstp = (v16bf*)(Wpk + (size_t)t * 32);
  dstp[0] = hi;
  dstp[1] = lo;
}

// ---------------------------------------------------------------------------
// bf16 split-precision GEMM with WMMA: C[M x 128] = A[M x 128] @ W[128 x 128] + bias
// Block = 256 threads = 8 waves; each wave owns one 16x16 output tile.
// grid.x = M/128 (8 wave-tiles of 16 rows), grid.y = 8 (N tiles of 16 cols).
// B fragments come pre-packed in bf16 (no conversion in the hot loop).
// ntStore=1 -> nontemporal C stores (streaming output, keep L2 for q/k/v).
// ---------------------------------------------------------------------------
__global__ void __launch_bounds__(256)
gemm_bias_kernel(const float* __restrict__ A, const __bf16* __restrict__ Wpk,
                 const float* __restrict__ bias, float* __restrict__ C, int M,
                 int ntStore) {
  const int wave = threadIdx.x >> 5;
  const int lane = threadIdx.x & 31;
  const int h = lane >> 4;        // lane half (0/1)
  const int r = lane & 15;
  const int mbase = blockIdx.x * 128 + wave * 16;
  const int nbase = blockIdx.y * 16;
  const int ncol  = nbase + r;    // this lane's column for C/D
  const float* __restrict__ arow = A + (size_t)(mbase + r) * Dc;

  v8f acc = {};
#pragma unroll
  for (int kbi = 0; kbi < 4; ++kbi) {
    const int kb = kbi * 32;
    // ---- A fragment (16x32 bf16), contiguous k order:
    // lane-half h, element j holds K = kb + h*16 + j (matches packed B)
    // => one contiguous 16-float run per lane: [kb+h*16, +16)
    v16bf ah, al;
#pragma unroll
    for (int c = 0; c < 2; ++c) {
      const float* p = arow + kb + h * 16 + 8 * c;
#pragma unroll
      for (int j = 0; j < 8; ++j) {
        float xv = p[j];
        __bf16 hib = (__bf16)xv;
        ah[8 * c + j] = hib;
        al[8 * c + j] = (__bf16)(xv - (float)hib);
      }
    }
    // ---- B fragment: pre-packed, two contiguous 32B vector loads
    const v16bf* wp =
        (const v16bf*)(Wpk + (((size_t)blockIdx.y * 4 + kbi) * 32 + lane) * 32);
    v16bf bh = wp[0];
    v16bf bl = wp[1];
    // split-precision product: Ah*Bh + Ah*Bl + Al*Bh  (~fp32 accuracy)
    acc = __builtin_amdgcn_wmma_f32_16x16x32_bf16(false, ah, false, bh, (short)0, acc, false, false);
    acc = __builtin_amdgcn_wmma_f32_16x16x32_bf16(false, ah, false, bl, (short)0, acc, false, false);
    acc = __builtin_amdgcn_wmma_f32_16x16x32_bf16(false, al, false, bh, (short)0, acc, false, false);
  }
  const float bv = bias[ncol];
  // C/D layout: VGPR i -> (M = mbase + i + 8*h, N = ncol)
  if (ntStore) {
#pragma unroll
    for (int i = 0; i < 8; ++i)
      __builtin_nontemporal_store(acc[i] + bv, &C[(size_t)(mbase + i + 8 * h) * Dc + ncol]);
  } else {
#pragma unroll
    for (int i = 0; i < 8; ++i)
      C[(size_t)(mbase + i + 8 * h) * Dc + ncol] = acc[i] + bv;
  }
}

// ---------------------------------------------------------------------------
// Wave helpers (wave32)
// ---------------------------------------------------------------------------
__device__ __forceinline__ float wave_sum(float v) {
#pragma unroll
  for (int off = 16; off > 0; off >>= 1) v += __shfl_xor(v, off, 32);
  return v;
}
__device__ __forceinline__ unsigned enc_mono(float f) {
  unsigned u = __float_as_uint(f);
  return (u & 0x80000000u) ? ~u : (u | 0x80000000u);
}
__device__ __forceinline__ float dec_mono(unsigned e) {
  unsigned u = (e & 0x80000000u) ? (e & 0x7FFFFFFFu) : ~e;
  return __uint_as_float(u);
}

// ---------------------------------------------------------------------------
// alpha = dot(q[dst], k[src]+e) / sqrt(D); segment-max via monotone atomicMax
// One wave per edge; lane covers 4 dims (float4). q/k are L2-resident (32MB).
// ---------------------------------------------------------------------------
__global__ void __launch_bounds__(256)
edge_alpha_kernel(const int* __restrict__ src, const int* __restrict__ dst,
                  const float* __restrict__ q, const float* __restrict__ k,
                  const float* __restrict__ e, float* __restrict__ alpha,
                  unsigned* __restrict__ menc) {
  const long long eid = (long long)blockIdx.x * 8 + (threadIdx.x >> 5);
  const int lane = threadIdx.x & 31;
  if (eid >= Ec) return;
  const int s = src[eid], d = dst[eid];
  float4 qv = ((const float4*)q)[(size_t)d * 32 + lane];
  float4 kv = ((const float4*)k)[(size_t)s * 32 + lane];
  float4 ev = ((const float4*)e)[(size_t)eid * 32 + lane];
  float p = qv.x * (kv.x + ev.x) + qv.y * (kv.y + ev.y) +
            qv.z * (kv.z + ev.z) + qv.w * (kv.w + ev.w);
  p = wave_sum(p);
  if (lane == 0) {
    float a = p * 0.0883883476483184f;  // 1/sqrt(128)
    alpha[eid] = a;
    atomicMax(&menc[d], enc_mono(a));
  }
}

// ex = exp(alpha - m[dst]); denom[dst] += ex
__global__ void edge_exp_kernel(const int* __restrict__ dst,
                                const float* __restrict__ alpha,
                                const unsigned* __restrict__ menc,
                                float* __restrict__ exb, float* __restrict__ denom) {
  const long long eid = (long long)blockIdx.x * blockDim.x + threadIdx.x;
  if (eid >= Ec) return;
  const int d = dst[eid];
  float ex = __expf(alpha[eid] - dec_mono(menc[d]));
  exb[eid] = ex;
  atomicAdd(&denom[d], ex);
}

// out[dst] += attn * (v[src] + e)   (L2-resident fp32 atomics; out is 32 MB)
__global__ void __launch_bounds__(256)
edge_msg_kernel(const int* __restrict__ src, const int* __restrict__ dst,
                const float* __restrict__ v, const float* __restrict__ e,
                const float* __restrict__ exb, const float* __restrict__ denom,
                float* __restrict__ outb) {
  const long long eid = (long long)blockIdx.x * 8 + (threadIdx.x >> 5);
  const int lane = threadIdx.x & 31;
  if (eid >= Ec) return;
  const int s = src[eid], d = dst[eid];
  const float attn = exb[eid] / (denom[d] + 1e-16f);
  float4 vv = ((const float4*)v)[(size_t)s * 32 + lane];
  float4 ev = ((const float4*)e)[(size_t)eid * 32 + lane];
  float* base = outb + (size_t)d * Dc + lane * 4;
  atomicAdd(base + 0, attn * (vv.x + ev.x));
  atomicAdd(base + 1, attn * (vv.y + ev.y));
  atomicAdd(base + 2, attn * (vv.z + ev.z));
  atomicAdd(base + 3, attn * (vv.w + ev.w));
}

// 1/||pool_w||  (single wave)
__global__ void pool_norm_kernel(const float* __restrict__ pw, float* __restrict__ scale) {
  const int lane = threadIdx.x & 31;
  float4 w = ((const float4*)pw)[lane];
  float p = w.x * w.x + w.y * w.y + w.z * w.z + w.w * w.w;
  p = wave_sum(p);
  if (lane == 0) scale[0] = rsqrtf(p);
}

// out = relu(out + skip); score = tanh(dot(out, pw) / ||pw||). One wave per node.
__global__ void __launch_bounds__(256)
node_relu_score_kernel(float* __restrict__ outb, const float* __restrict__ sk,
                       const float* __restrict__ pw, const float* __restrict__ pscale,
                       float* __restrict__ score) {
  const int nid = blockIdx.x * 8 + (threadIdx.x >> 5);
  const int lane = threadIdx.x & 31;
  if (nid >= NNc) return;
  float4 o = ((float4*)outb)[(size_t)nid * 32 + lane];
  float4 s = ((const float4*)sk)[(size_t)nid * 32 + lane];
  o.x = fmaxf(o.x + s.x, 0.f); o.y = fmaxf(o.y + s.y, 0.f);
  o.z = fmaxf(o.z + s.z, 0.f); o.w = fmaxf(o.w + s.w, 0.f);
  ((float4*)outb)[(size_t)nid * 32 + lane] = o;
  float4 w = ((const float4*)pw)[lane];
  float p = o.x * w.x + o.y * w.y + o.z * w.z + o.w * w.w;
  p = wave_sum(p);
  if (lane == 0) score[nid] = tanhf(p * pscale[0]);
}

// ---------------------------------------------------------------------------
// Per-graph sorted top-K: bitonic sort of 1024 (score,idx) pairs in LDS,
// descending score, ascending index on ties (matches jax.lax.top_k ordering).
// One 1024-thread workgroup per graph.
// ---------------------------------------------------------------------------
__global__ void __launch_bounds__(1024)
topk_kernel(const float* __restrict__ score, float* __restrict__ tops,
            int* __restrict__ perm, int* __restrict__ node_map,
            float* __restrict__ out_batch) {
  __shared__ float ss[Nc];
  __shared__ int   ii[Nc];
  const int g = blockIdx.x;
  const int t = threadIdx.x;
  ss[t] = score[(size_t)g * Nc + t];
  ii[t] = t;
  __syncthreads();
  for (int k2 = 2; k2 <= Nc; k2 <<= 1) {
    for (int j = k2 >> 1; j > 0; j >>= 1) {
      const int ixj = t ^ j;
      if (ixj > t) {
        float st = ss[t], sx = ss[ixj];
        int   it = ii[t], ix = ii[ixj];
        bool tWorse = (st < sx) || (st == sx && it > ix);  // later in best-first order
        bool bestFirst = ((t & k2) == 0);
        if (bestFirst ? tWorse : !tWorse) {
          ss[t] = sx; ss[ixj] = st; ii[t] = ix; ii[ixj] = it;
        }
      }
      __syncthreads();
    }
  }
  if (t < Kc) {
    const int row = g * Kc + t;
    const int gid = g * Nc + ii[t];
    perm[row] = gid;
    tops[row] = ss[t];
    node_map[gid] = row;
    out_batch[row] = (float)g;
  }
}

// xp[row] = out[perm[row]] * tops[row]   (one wave per kept row)
__global__ void __launch_bounds__(256)
xp_kernel(const float* __restrict__ outb, const int* __restrict__ perm,
          const float* __restrict__ tops, float* __restrict__ xp) {
  const int row = blockIdx.x * 8 + (threadIdx.x >> 5);
  const int lane = threadIdx.x & 31;
  if (row >= BKc) return;
  const int p = perm[row];
  const float s = tops[row];
  float4 o = ((const float4*)outb)[(size_t)p * 32 + lane];
  o.x *= s; o.y *= s; o.z *= s; o.w *= s;
  ((float4*)xp)[(size_t)row * 32 + lane] = o;
}

// Per-column sums for BN: 128 blocks x 128 threads, each block covers 256 rows.
__global__ void __launch_bounds__(128)
bn_stats_kernel(const float* __restrict__ xp, float* __restrict__ colsum,
                float* __restrict__ colsq) {
  const int col = threadIdx.x;
  const int r0 = blockIdx.x * 256;
  float sum = 0.f, sq = 0.f;
  for (int r = r0; r < r0 + 256; ++r) {
    float v = xp[(size_t)r * Dc + col];
    sum += v; sq += v * v;
  }
  atomicAdd(&colsum[col], sum);
  atomicAdd(&colsq[col], sq);
}

__global__ void bn_apply_kernel(const float* __restrict__ xp, const float* __restrict__ colsum,
                                const float* __restrict__ colsq, const float* __restrict__ gamma,
                                const float* __restrict__ beta, float* __restrict__ y) {
  const size_t idx = (size_t)blockIdx.x * blockDim.x + threadIdx.x;
  if (idx >= (size_t)BKc * Dc) return;
  const int col = idx & (Dc - 1);
  const float inv = 1.0f / (float)BKc;
  float mu = colsum[col] * inv;
  float var = colsq[col] * inv - mu * mu;
  y[idx] = gamma[col] * (xp[idx] - mu) * rsqrtf(var + 1e-5f) + beta[col];
}

// Edge remap/filter: writes new_edge_index, new_edge_attr, valid (as floats).
// new_edge_attr is a 512 MB pure stream -> nontemporal stores (native vec type).
__global__ void __launch_bounds__(256)
edge_filter_kernel(const int* __restrict__ src, const int* __restrict__ dst,
                   const int* __restrict__ node_map, const float* __restrict__ ea,
                   float* __restrict__ out_ei, float* __restrict__ out_ea,
                   float* __restrict__ out_valid) {
  const long long eid = (long long)blockIdx.x * 8 + (threadIdx.x >> 5);
  const int lane = threadIdx.x & 31;
  if (eid >= Ec) return;
  const int ns = node_map[src[eid]];
  const int nd = node_map[dst[eid]];
  const bool val = (ns >= 0) && (nd >= 0);
  const float f = val ? 1.0f : 0.0f;
  f32x4 a = ((const f32x4*)ea)[(size_t)eid * 32 + lane];
  a *= f;
  __builtin_nontemporal_store(a, &((f32x4*)out_ea)[(size_t)eid * 32 + lane]);
  if (lane == 0) {
    out_ei[eid]      = val ? (float)ns : -1.0f;
    out_ei[Ec + eid] = val ? (float)nd : -1.0f;
    out_valid[eid] = f;
  }
}

// ---------------------------------------------------------------------------
extern "C" void kernel_launch(void* const* d_in, const int* in_sizes, int n_in,
                              void* d_out, int out_size, void* d_ws, size_t ws_size,
                              hipStream_t stream) {
  const float* x   = (const float*)d_in[0];
  const int*   ei  = (const int*)d_in[1];
  const float* ea  = (const float*)d_in[2];
  const float* Wq  = (const float*)d_in[4];  const float* bq = (const float*)d_in[5];
  const float* Wk  = (const float*)d_in[6];  const float* bk = (const float*)d_in[7];
  const float* Wv  = (const float*)d_in[8];  const float* bv = (const float*)d_in[9];
  const float* We  = (const float*)d_in[10]; const float* be = (const float*)d_in[11];
  const float* Wsk = (const float*)d_in[12]; const float* bsk = (const float*)d_in[13];
  const float* pw  = (const float*)d_in[14];
  const float* gamma = (const float*)d_in[15];
  const float* beta  = (const float*)d_in[16];
  const int* src = ei;
  const int* dst = ei + Ec;

  // Workspace layout (floats)
  float* ws = (float*)d_ws;
  float* q   = ws;                         // NN*D
  float* kf  = q  + (size_t)NNc * Dc;      // NN*D
  float* vf  = kf + (size_t)NNc * Dc;      // NN*D
  float* sk  = vf + (size_t)NNc * Dc;      // NN*D
  float* ef  = sk + (size_t)NNc * Dc;      // E*D  (512 MiB)
  float* alpha = ef + (size_t)Ec * Dc;     // E
  float* exb   = alpha + Ec;               // E
  unsigned* menc = (unsigned*)(exb + Ec);  // NN
  float* denom = (float*)(menc + NNc);     // NN
  float* outb  = denom + NNc;              // NN*D
  float* score = outb + (size_t)NNc * Dc;  // NN
  float* tops  = score + NNc;              // B*K
  int*   perm  = (int*)(tops + BKc);       // B*K
  int*   nodemap = perm + BKc;             // NN
  float* xp    = (float*)(nodemap + NNc);  // B*K*D
  float* colsum = xp + (size_t)BKc * Dc;   // 128
  float* colsq  = colsum + Dc;             // 128
  float* pscale = colsq + Dc;              // 1
  // packed bf16 weights (32B-aligned offset)
  size_t off = (size_t)((pscale + 1) - ws);
  off = (off + 15) & ~(size_t)15;
  __bf16* wpkq = (__bf16*)(ws + off);      // each matrix: WPK_ELEMS bf16 (64 KB)
  __bf16* wpkk = wpkq + WPK_ELEMS;
  __bf16* wpkv = wpkk + WPK_ELEMS;
  __bf16* wpks = wpkv + WPK_ELEMS;
  __bf16* wpke = wpks + WPK_ELEMS;

  // Output layout (floats, concatenated in reference return order)
  float* out_y     = (float*)d_out;              // [B*K, D]
  float* out_ei    = out_y  + (size_t)BKc * Dc;  // [2, E]
  float* out_ea    = out_ei + 2 * (size_t)Ec;    // [E, D]
  float* out_batch = out_ea + (size_t)Ec * Dc;   // [B*K]
  float* out_valid = out_batch + BKc;            // [E]

  // ---- init scratch -------------------------------------------------------
  {
    size_t n = (size_t)NNc * Dc;
    fill_f32_kernel<<<(unsigned)((n + 255) / 256), 256, 0, stream>>>(outb, 0.f, n);
    fill_f32_kernel<<<(NNc + 255) / 256, 256, 0, stream>>>(denom, 0.f, NNc);
    fill_u32_kernel<<<(NNc + 255) / 256, 256, 0, stream>>>(menc, 0u, NNc);
    fill_u32_kernel<<<(NNc + 255) / 256, 256, 0, stream>>>((unsigned*)nodemap, 0xFFFFFFFFu, NNc);
    fill_f32_kernel<<<1, 256, 0, stream>>>(colsum, 0.f, 2 * Dc + 1); // colsum+colsq (+pad)
  }
  pool_norm_kernel<<<1, 32, 0, stream>>>(pw, pscale);

  // ---- pre-pack weights to fragment-ordered bf16 hi/lo --------------------
  pack_w_kernel<<<4, 256, 0, stream>>>(Wq, wpkq);
  pack_w_kernel<<<4, 256, 0, stream>>>(Wk, wpkk);
  pack_w_kernel<<<4, 256, 0, stream>>>(Wv, wpkv);
  pack_w_kernel<<<4, 256, 0, stream>>>(Wsk, wpks);
  pack_w_kernel<<<4, 256, 0, stream>>>(We, wpke);

  // ---- GEMMs (WMMA bf16 split-precision) ---------------------------------
  {
    dim3 blk(256);
    dim3 gN(NNc / 128, 8);
    gemm_bias_kernel<<<gN, blk, 0, stream>>>(x, wpkq, bq, q, NNc, 0);
    gemm_bias_kernel<<<gN, blk, 0, stream>>>(x, wpkk, bk, kf, NNc, 0);
    gemm_bias_kernel<<<gN, blk, 0, stream>>>(x, wpkv, bv, vf, NNc, 0);
    gemm_bias_kernel<<<gN, blk, 0, stream>>>(x, wpks, bsk, sk, NNc, 0);
    dim3 gE((unsigned)(Ec / 128), 8);
    gemm_bias_kernel<<<gE, blk, 0, stream>>>(ea, wpke, be, ef, (int)Ec, 1);  // NT stores
  }

  // ---- attention ----------------------------------------------------------
  const unsigned ewaves = (unsigned)(Ec / 8);
  edge_alpha_kernel<<<ewaves, 256, 0, stream>>>(src, dst, q, kf, ef, alpha, menc);
  edge_exp_kernel<<<(unsigned)((Ec + 255) / 256), 256, 0, stream>>>(dst, alpha, menc, exb, denom);
  edge_msg_kernel<<<ewaves, 256, 0, stream>>>(src, dst, vf, ef, exb, denom, outb);
  node_relu_score_kernel<<<NNc / 8, 256, 0, stream>>>(outb, sk, pw, pscale, score);

  // ---- top-K pooling ------------------------------------------------------
  topk_kernel<<<Bc, 1024, 0, stream>>>(score, tops, perm, nodemap, out_batch);
  xp_kernel<<<BKc / 8, 256, 0, stream>>>(outb, perm, tops, xp);

  // ---- batch norm ---------------------------------------------------------
  bn_stats_kernel<<<BKc / 256, 128, 0, stream>>>(xp, colsum, colsq);
  bn_apply_kernel<<<(unsigned)(((size_t)BKc * Dc + 255) / 256), 256, 0, stream>>>(
      xp, colsum, colsq, gamma, beta, out_y);

  // ---- edge remap / filter ------------------------------------------------
  edge_filter_kernel<<<ewaves, 256, 0, stream>>>(src, dst, nodemap, ea, out_ei, out_ea, out_valid);
}
